// PointTrans_Layer_down_23673859735699
// MI455X (gfx1250) — matrix operand order repaired and checked by reading, not compiled, not run on hardware
//
#include <hip/hip_runtime.h>
#include <hip/hip_bf16.h>

typedef __attribute__((ext_vector_type(16))) _Float16 v16h;
typedef __attribute__((ext_vector_type(8)))  _Float16 v8h;
typedef __attribute__((ext_vector_type(8)))  float    v8f;

#define N_PTS   32768
#define C_INF   128
#define C_OUTF  128
#define N_EDGE  524288
#define N_SAMP  16384
#define BN_EPS  1e-5f

static __device__ __forceinline__ v16h cat8(v8h a, v8h b) {
    v16h r;
    #pragma unroll
    for (int j = 0; j < 8; ++j) { r[j] = a[j]; r[j + 8] = b[j]; }
    return r;
}

// ---------------------------------------------------------------------------
// Kernel 0: split fp32 -> (f16 hi, f16 lo) so the GEMM hot loop has zero
// conversion VALU work:  f = hi + lo exactly captures ~f32 precision via
// three f16 WMMA products (hi*hi + hi*lo + lo*hi).
// ---------------------------------------------------------------------------
__global__ __launch_bounds__(256)
void split_f16_kernel(const float* __restrict__ src, _Float16* __restrict__ hi,
                      _Float16* __restrict__ lo, int n) {
    int i = blockIdx.x * 256 + threadIdx.x;
    if (i >= n) return;
    float f = src[i];
    _Float16 hh = (_Float16)f;
    hi[i] = hh;
    lo[i] = (_Float16)(f - (float)hh);
}

// ---------------------------------------------------------------------------
// Kernel 1: h = x @ W^T + b via V_WMMA_F32_16X16X32_F16.
// One wave computes a full 16x128 output strip (8 N-tiles): each A fragment
// is reused by 24 WMMAs; 96 WMMAs per wave total. 4 waves / block.
// Fragment layouts used here (contiguous k order, same for A and B):
//   A (16x32): lane row M=L%16; halves j   -> K=hs*16+j
//   B (32x16): lane col N=L%16; halves j   -> K=hs*16+j
//   D: VGPR r -> row hs*8+r, col L%16
// ---------------------------------------------------------------------------
__global__ __launch_bounds__(128)
void gemm_wmma_kernel(const _Float16* __restrict__ xh, const _Float16* __restrict__ xl,
                      const _Float16* __restrict__ wh, const _Float16* __restrict__ wl,
                      const float* __restrict__ bias, float* __restrict__ h) {
    const int wid   = threadIdx.x >> 5;
    const int lane  = threadIdx.x & 31;
    const int hs    = lane >> 4;
    const int l16   = lane & 15;
    const int mtile = blockIdx.x * 4 + wid;      // 0..2047
    const int mrow  = mtile * 16 + l16;

    const _Float16* __restrict__ xh_row = xh + (size_t)mrow * C_INF;
    const _Float16* __restrict__ xl_row = xl + (size_t)mrow * C_INF;

    v8f acc[8];
    #pragma unroll
    for (int nt = 0; nt < 8; ++nt) acc[nt] = (v8f){};

    #pragma unroll
    for (int kk = 0; kk < C_INF; kk += 32) {
        if (kk + 32 < C_INF) {
            __builtin_prefetch(xh_row + kk + 32, 0, 0);   // global_prefetch_b8
            __builtin_prefetch(xl_row + kk + 32, 0, 0);
        }
        v16h a_hi = cat8(*(const v8h*)(xh_row + kk + hs * 16),
                         *(const v8h*)(xh_row + kk + hs * 16 + 8));
        v16h a_lo = cat8(*(const v8h*)(xl_row + kk + hs * 16),
                         *(const v8h*)(xl_row + kk + hs * 16 + 8));
        #pragma unroll
        for (int nt = 0; nt < 8; ++nt) {
            const size_t boff = (size_t)(nt * 16 + l16) * C_INF + kk + hs * 16;
            v16h b_hi = *(const v16h*)(wh + boff);   // 32B aligned contiguous
            v16h b_lo = *(const v16h*)(wl + boff);
            acc[nt] = __builtin_amdgcn_wmma_f32_16x16x32_f16(false, a_hi, false, b_hi,
                                                             (short)0, acc[nt], false, false);
            acc[nt] = __builtin_amdgcn_wmma_f32_16x16x32_f16(false, a_hi, false, b_lo,
                                                             (short)0, acc[nt], false, false);
            acc[nt] = __builtin_amdgcn_wmma_f32_16x16x32_f16(false, a_lo, false, b_hi,
                                                             (short)0, acc[nt], false, false);
        }
    }

    #pragma unroll
    for (int nt = 0; nt < 8; ++nt) {
        const int ncol = nt * 16 + l16;
        const float bv = bias[ncol];
        #pragma unroll
        for (int r = 0; r < 8; ++r) {
            int m = mtile * 16 + hs * 8 + r;
            h[(size_t)m * C_OUTF + ncol] = acc[nt][r] + bv;
        }
    }
}

// ---------------------------------------------------------------------------
// Kernel 2: per-channel batch mean / biased variance. One block per channel.
// ---------------------------------------------------------------------------
__global__ __launch_bounds__(256)
void stats_kernel(const float* __restrict__ h, float* __restrict__ mean,
                  float* __restrict__ var) {
    const int c = blockIdx.x;
    float s = 0.f, s2 = 0.f;
    for (int n = threadIdx.x; n < N_PTS; n += 256) {
        float v = h[(size_t)n * C_OUTF + c];
        s += v; s2 += v * v;
    }
    __shared__ float ls[256], ls2[256];
    ls[threadIdx.x] = s; ls2[threadIdx.x] = s2;
    __syncthreads();
    for (int off = 128; off > 0; off >>= 1) {
        if (threadIdx.x < off) {
            ls[threadIdx.x]  += ls[threadIdx.x + off];
            ls2[threadIdx.x] += ls2[threadIdx.x + off];
        }
        __syncthreads();
    }
    if (threadIdx.x == 0) {
        float m = ls[0] / (float)N_PTS;
        mean[c] = m;
        var[c]  = ls2[0] / (float)N_PTS - m * m;
    }
}

// ---------------------------------------------------------------------------
// Kernel 3: BN (training stats) + ReLU; writes normalized h and pooled = h.
// ---------------------------------------------------------------------------
__global__ __launch_bounds__(256)
void bnrelu_kernel(float* __restrict__ h, float* __restrict__ pooled,
                   const float* __restrict__ mean, const float* __restrict__ var,
                   const float* __restrict__ gamma, const float* __restrict__ beta) {
    size_t i = (size_t)blockIdx.x * 256 + threadIdx.x;
    if (i >= (size_t)N_PTS * C_OUTF) return;
    int c = (int)(i & (C_OUTF - 1));
    float v = (h[i] - mean[c]) * rsqrtf(var[c] + BN_EPS) * gamma[c] + beta[c];
    v = fmaxf(v, 0.f);
    h[i] = v;
    pooled[i] = v;
}

// ---------------------------------------------------------------------------
// Kernel 4: scatter-max over edges. One wave32 per edge, 4 channels per lane.
// Post-ReLU values >= 0, so float max == signed-int max on bit patterns.
// ---------------------------------------------------------------------------
__global__ __launch_bounds__(256)
void edge_max_kernel(const int* __restrict__ edges, const float* __restrict__ h,
                     float* __restrict__ pooled) {
    int wave = blockIdx.x * (256 / 32) + (threadIdx.x >> 5);
    int lane = threadIdx.x & 31;
    if (wave >= N_EDGE) return;
    int r = edges[wave];
    int c = edges[N_EDGE + wave];
    const float* src = h + (size_t)r * C_OUTF;
    int* dst = (int*)(pooled + (size_t)c * C_OUTF);
    #pragma unroll
    for (int j = 0; j < 4; ++j) {
        int ch = lane + 32 * j;
        atomicMax(dst + ch, __float_as_int(src[ch]));
    }
}

// ---------------------------------------------------------------------------
// Kernel 5: farthest point sampling, single persistent workgroup. min_d lives
// in 139 KB of dynamic LDS (CDNA5 WGP has 320 KB), removing the 256 KB global
// round-trip from each of the 16383 sequential iterations.
// ---------------------------------------------------------------------------
__global__ __launch_bounds__(1024)
void fps_kernel(const float* __restrict__ pos, int* __restrict__ flags) {
    extern __shared__ float smem[];
    float* min_d = smem;                              // [N_PTS]
    float* rv    = smem + N_PTS;                      // [1024]
    int*   ri    = (int*)(smem + N_PTS + 1024);       // [1024]
    int*   sbest = (int*)(smem + N_PTS + 2048);       // [1]
    const int t = threadIdx.x;

    float x0 = pos[0], y0 = pos[1], z0 = pos[2];
    for (int i = t; i < N_PTS; i += 1024) {
        float dx = pos[i*3+0]-x0, dy = pos[i*3+1]-y0, dz = pos[i*3+2]-z0;
        min_d[i] = dx*dx + dy*dy + dz*dz;
        flags[i] = (i == 0) ? 1 : 0;
    }
    __syncthreads();

    for (int it = 0; it < N_SAMP - 1; ++it) {
        // local argmax (ascending indices -> strict > keeps first occurrence)
        float bv = -1.0f; int bi = 0x7fffffff;
        for (int i = t; i < N_PTS; i += 1024) {
            float v = min_d[i];
            if (v > bv) { bv = v; bi = i; }
        }
        rv[t] = bv; ri[t] = bi;
        __syncthreads();
        for (int off = 512; off > 0; off >>= 1) {
            if (t < off) {
                float v2 = rv[t+off]; int i2 = ri[t+off];
                if (v2 > rv[t] || (v2 == rv[t] && i2 < ri[t])) { rv[t] = v2; ri[t] = i2; }
            }
            __syncthreads();
        }
        if (t == 0) { *sbest = ri[0]; flags[ri[0]] = 1; }
        __syncthreads();
        int best = *sbest;
        float bx = pos[best*3+0], by = pos[best*3+1], bz = pos[best*3+2];
        for (int i = t; i < N_PTS; i += 1024) {
            float dx = pos[i*3+0]-bx, dy = pos[i*3+1]-by, dz = pos[i*3+2]-bz;
            float d = dx*dx + dy*dy + dz*dz;
            float o = min_d[i];
            min_d[i] = d < o ? d : o;
        }
        __syncthreads();
    }
}

// ---------------------------------------------------------------------------
// Kernel 6: order-preserving compaction of selection bitmap -> sorted indices.
// ---------------------------------------------------------------------------
__global__ __launch_bounds__(1024)
void compact_kernel(const int* __restrict__ flags, int* __restrict__ idx_list) {
    const int t = threadIdx.x;
    __shared__ int cnt[1024];
    int base = t * 32;
    int c = 0;
    for (int j = 0; j < 32; ++j) c += flags[base + j];
    cnt[t] = c;
    __syncthreads();
    for (int off = 1; off < 1024; off <<= 1) {
        int v = (t >= off) ? cnt[t - off] : 0;
        __syncthreads();
        cnt[t] += v;
        __syncthreads();
    }
    int p = cnt[t] - c;   // exclusive prefix
    for (int j = 0; j < 32; ++j) {
        if (flags[base + j]) idx_list[p++] = base + j;
    }
}

// ---------------------------------------------------------------------------
// Kernel 7: gather outputs pooled[idx], pos[idx], batch[idx], y[idx] (flat).
// ---------------------------------------------------------------------------
__global__ __launch_bounds__(128)
void gather_kernel(const int* __restrict__ idx_list, const float* __restrict__ pooled,
                   const float* __restrict__ pos, const int* __restrict__ batch,
                   const int* __restrict__ y, float* __restrict__ out) {
    int i = blockIdx.x;
    int t = threadIdx.x;
    int n = idx_list[i];
    const size_t O1 = (size_t)N_SAMP * C_OUTF;
    const size_t O2 = O1 + (size_t)N_SAMP * 3;
    const size_t O3 = O2 + (size_t)N_SAMP;
    out[(size_t)i * C_OUTF + t] = pooled[(size_t)n * C_OUTF + t];
    if (t < 3) out[O1 + (size_t)i * 3 + t] = pos[(size_t)n * 3 + t];
    if (t == 0) {
        out[O2 + i] = (float)batch[n];
        out[O3 + i] = (float)y[n];
    }
}

// ---------------------------------------------------------------------------
extern "C" void kernel_launch(void* const* d_in, const int* in_sizes, int n_in,
                              void* d_out, int out_size, void* d_ws, size_t ws_size,
                              hipStream_t stream) {
    const float* x     = (const float*)d_in[0];
    const float* pos   = (const float*)d_in[1];
    const int*   edges = (const int*)  d_in[2];
    const int*   batch = (const int*)  d_in[3];
    const int*   y     = (const int*)  d_in[4];
    const float* W     = (const float*)d_in[5];
    const float* bias  = (const float*)d_in[6];
    const float* gamma = (const float*)d_in[7];
    const float* beta  = (const float*)d_in[8];
    float* out = (float*)d_out;

    // Workspace layout (xh/xl alias the pooled buffer: dead until bnrelu).
    char* ws = (char*)d_ws;
    float*     h      = (float*)    (ws);                          // 16 MB
    float*     pooled = (float*)    (ws + (size_t)16777216);       // 16 MB
    _Float16*  xh     = (_Float16*) (ws + (size_t)16777216);       // 8 MB (alias)
    _Float16*  xl     = (_Float16*) (ws + (size_t)25165824);       // 8 MB (alias)
    _Float16*  wh     = (_Float16*) (ws + (size_t)33554432);       // 32 KB
    _Float16*  wl     = (_Float16*) (ws + (size_t)33587200);       // 32 KB
    float*     mean   = (float*)    (ws + (size_t)33619968);       // 512 B
    float*     var    = (float*)    (ws + (size_t)33620480);       // 512 B
    int*       flags  = (int*)      (ws + (size_t)33620992);       // 128 KB
    int*       idxl   = (int*)      (ws + (size_t)33752064);       // 64 KB

    split_f16_kernel<<<(N_PTS * C_INF) / 256, 256, 0, stream>>>(x, xh, xl, N_PTS * C_INF);
    split_f16_kernel<<<(C_OUTF * C_INF) / 256, 256, 0, stream>>>(W, wh, wl, C_OUTF * C_INF);

    gemm_wmma_kernel<<<512, 128, 0, stream>>>(xh, xl, wh, wl, bias, h);

    stats_kernel<<<128, 256, 0, stream>>>(h, mean, var);
    bnrelu_kernel<<<((size_t)N_PTS * C_OUTF) / 256, 256, 0, stream>>>(
        h, pooled, mean, var, gamma, beta);
    edge_max_kernel<<<N_EDGE / 8, 256, 0, stream>>>(edges, h, pooled);

    const int fps_lds = (N_PTS + 1024 + 1024 + 16) * sizeof(float);  // ~139 KB dynamic LDS
    fps_kernel<<<1, 1024, fps_lds, stream>>>(pos, flags);

    compact_kernel<<<1, 1024, 0, stream>>>(flags, idxl);
    gather_kernel<<<N_SAMP, 128, 0, stream>>>(idxl, pooled, pos, batch, y, out);
}
